// FilteredConvMixer_14740327759891
// MI455X (gfx1250) — matrix operand order repaired and checked by reading, not compiled, not run on hardware
//
#include <hip/hip_runtime.h>
#include <hip/hip_bf16.h>

typedef __attribute__((ext_vector_type(16))) _Float16 v16h;
typedef __attribute__((ext_vector_type(8)))  _Float16 v8h;
typedef __attribute__((ext_vector_type(8)))  float    v8f;

#define BATCH 16
#define IH 224
#define IW 224
#define HWPIX (IH*IW)        // 50176
#define HID 64
#define DIM 256
#define DEPTH 8
#define GH 32
#define GW 32
#define GHW 1024
#define NCLS 1000
#define NBINS 2048
#define KSTEM 160            // stem K padded 147 -> 160

__device__ __forceinline__ float gelu_exact(float x) {
    return 0.5f * x * (1.0f + erff(x * 0.70710678118654752f));
}

__device__ __forceinline__ v16h cat16(v8h a, v8h b) {
    return __builtin_shufflevector(a, b, 0,1,2,3,4,5,6,7,8,9,10,11,12,13,14,15);
}

// ---------------- filter CNN: conv1 3->64 3x3 SAME, NCHW in -> NHWC f32 raw ----------------
__global__ void conv1_kernel(const float* __restrict__ x, const float* __restrict__ w,
                             float* __restrict__ raw) {
    int idx = blockIdx.x * blockDim.x + threadIdx.x;   // B*HW*64
    if (idx >= BATCH * HWPIX * HID) return;
    int co = idx & 63;
    int pix = idx >> 6;
    int b = pix / HWPIX; int p = pix - b * HWPIX;
    int y = p / IW; int xw = p - y * IW;
    float acc = 0.f;
    #pragma unroll
    for (int ci = 0; ci < 3; ++ci)
        for (int ky = 0; ky < 3; ++ky) {
            int yy = y + ky - 1; if ((unsigned)yy >= IH) continue;
            for (int kx = 0; kx < 3; ++kx) {
                int xx = xw + kx - 1; if ((unsigned)xx >= IW) continue;
                acc += x[((size_t)(b * 3 + ci) * IH + yy) * IW + xx]
                     * w[((co * 3 + ci) * 3 + ky) * 3 + kx];
            }
        }
    raw[idx] = acc;
}

// ---------------- per-channel batch stats over NHWC [rows, C] ----------------
__global__ void stats_kernel(const float* __restrict__ v, int rows, int C,
                             float* __restrict__ mean, float* __restrict__ var) {
    int c = blockIdx.x;
    float s = 0.f, s2 = 0.f;
    for (int r = threadIdx.x; r < rows; r += blockDim.x) {
        float t = v[(size_t)r * C + c]; s += t; s2 += t * t;
    }
    __shared__ float sh[256], sh2[256];
    sh[threadIdx.x] = s; sh2[threadIdx.x] = s2; __syncthreads();
    for (int st = 128; st > 0; st >>= 1) {
        if ((int)threadIdx.x < st) { sh[threadIdx.x] += sh[threadIdx.x + st];
                                     sh2[threadIdx.x] += sh2[threadIdx.x + st]; }
        __syncthreads();
    }
    if (threadIdx.x == 0) {
        float m = sh[0] / (float)rows;
        mean[c] = m; var[c] = sh2[0] / (float)rows - m * m;
    }
}

// ---------------- BN + ReLU -> f16 activations (filter path) ----------------
__global__ void bn_relu_f16_kernel(const float* __restrict__ raw,
                                   const float* __restrict__ mean, const float* __restrict__ var,
                                   const float* __restrict__ g, const float* __restrict__ be,
                                   _Float16* __restrict__ act, int rows, int C) {
    size_t idx = (size_t)blockIdx.x * blockDim.x + threadIdx.x;
    if (idx >= (size_t)rows * C) return;
    int c = (int)(idx % C);
    float t = g[c] * (raw[idx] - mean[c]) * rsqrtf(var[c] + 1e-5f) + be[c];
    act[idx] = (_Float16)fmaxf(t, 0.f);
}

// ---------------- weight repack: f32 OIHW -> f16 TRANSPOSED [N, K] row-major ----------------
__global__ void wconv2T_kernel(const float* __restrict__ w, _Float16* __restrict__ BwT) {
    int i = blockIdx.x * blockDim.x + threadIdx.x;   // 64*576
    if (i >= 64 * 576) return;
    int o = i / 576; int k = i - o * 576;            // k = (ky*3+kx)*64 + c
    int c = k & 63; int t = k >> 6; int ky = t / 3, kx = t % 3;
    BwT[i] = (_Float16)w[((o * 64 + c) * 3 + ky) * 3 + kx];
}
__global__ void wstemT_kernel(const float* __restrict__ w, _Float16* __restrict__ BwT) {
    int i = blockIdx.x * blockDim.x + threadIdx.x;   // 256*160
    if (i >= 256 * KSTEM) return;
    int o = i / KSTEM; int k = i - o * KSTEM;        // k = c*49 + ky*7 + kx
    _Float16 v = (_Float16)0.f;
    if (k < 147) {
        int c = k / 49; int r = k - 49 * c; int ky = r / 7, kx = r % 7;
        v = (_Float16)w[((o * 3 + c) * 7 + ky) * 7 + kx];
    }
    BwT[i] = v;
}
__global__ void wpwT_kernel(const float* __restrict__ w, _Float16* __restrict__ BwT) {
    int i = blockIdx.x * blockDim.x + threadIdx.x;   // 256*256, already [out][in] = [N][K]
    if (i >= 256 * 256) return;
    BwT[i] = (_Float16)w[i];
}

// ============ LDS-free dense WMMA GEMM: C[M,N]f32 = A[M,K]f16 * BwT[N,K]f16^T ============
// One wave per 16x16 C tile. Fragments are two 8-half runs (kb, kb+16) per lane -> two b128 loads each.
__global__ __launch_bounds__(256)
void gemm_dense_wmma(const _Float16* __restrict__ A, const _Float16* __restrict__ BwT,
                     float* __restrict__ C, int M, int N, int K) {
    const int lane = threadIdx.x & 31;
    const int wid = blockIdx.x * 8 + (threadIdx.x >> 5);
    const int ntiles = N >> 4;
    const int tm = (wid / ntiles) << 4;
    const int tn = (wid % ntiles) << 4;
    if (tm >= M) return;
    const int l15 = lane & 15;
    const int kbA = (lane < 16) ? 0 : 8;    // A: halves kb..kb+7 and kb+16..kb+23 of row m
    const int kbB = (lane < 16) ? 0 : 8;    // B: halves kb..kb+7 and kb+16..kb+23 of row n
    const _Float16* arow = A + (size_t)(tm + l15) * K;
    const _Float16* brow = BwT + (size_t)(tn + l15) * K;
    v8f acc = {};
    for (int k0 = 0; k0 < K; k0 += 32) {
        v8h a0 = *(const v8h*)(arow + k0 + kbA);
        v8h a1 = *(const v8h*)(arow + k0 + kbA + 16);
        v8h b0 = *(const v8h*)(brow + k0 + kbB);
        v8h b1 = *(const v8h*)(brow + k0 + kbB + 16);
        acc = __builtin_amdgcn_wmma_f32_16x16x32_f16(false, cat16(a0, a1), false, cat16(b0, b1),
                                                     (short)0, acc, false, false);
    }
    const int n = tn + l15;
    const int mb = tm + ((lane < 16) ? 0 : 8);
    #pragma unroll
    for (int r = 0; r < 8; ++r)
        C[(size_t)(mb + r) * N + n] = acc[r];
}

// ============ conv2 (64->64, 3x3 SAME) implicit-im2col WMMA, LDS-free ============
// act: NHWC f16 [B,224,224,64]; BwT: [64][576] f16, k=(ky*3+kx)*64+c; C: NHWC f32.
// OOB handled branchlessly: clamp address, vector-select zero.
__global__ __launch_bounds__(256)
void gemm_conv2_wmma(const _Float16* __restrict__ act, const _Float16* __restrict__ BwT,
                     float* __restrict__ C) {
    const int K = 576, N = 64;
    const int lane = threadIdx.x & 31;
    const int wid = blockIdx.x * 8 + (threadIdx.x >> 5);
    const int tm = (wid >> 2) << 4;      // N/16 == 4 column tiles
    const int tn = (wid & 3) << 4;
    const int l15 = lane & 15;
    const int kbA = (lane < 16) ? 0 : 8;
    const int kbB = (lane < 16) ? 0 : 8;
    const int grow = tm + l15;           // this lane's output pixel
    const int b = grow / HWPIX; const int p = grow - b * HWPIX;
    const int y = p / IW; const int xw = p - y * IW;
    const _Float16* brow = BwT + (size_t)(tn + l15) * K;
    const v8h vzero = {};
    v8f acc = {};
    #pragma unroll
    for (int tap = 0; tap < 9; ++tap) {
        const int ky = tap / 3, kx = tap % 3;
        int yy = y + ky - 1, xx = xw + kx - 1;
        bool inb = ((unsigned)yy < IH) && ((unsigned)xx < IW);
        int yyc = inb ? yy : 0;
        int xxc = inb ? xx : 0;
        const _Float16* arow = act + (((size_t)(b * IH + yyc) * IW + xxc) << 6);
        #pragma unroll
        for (int half = 0; half < 2; ++half) {     // k0 = tap*64 + half*32
            const int cb = half * 32 + kbA;
            v8h a0 = *(const v8h*)(arow + cb);
            v8h a1 = *(const v8h*)(arow + cb + 16);
            a0 = inb ? a0 : vzero;
            a1 = inb ? a1 : vzero;
            const int k0 = tap * 64 + half * 32;
            v8h b0 = *(const v8h*)(brow + k0 + kbB);
            v8h b1 = *(const v8h*)(brow + k0 + kbB + 16);
            acc = __builtin_amdgcn_wmma_f32_16x16x32_f16(false, cat16(a0, a1), false,
                                                         cat16(b0, b1), (short)0, acc,
                                                         false, false);
        }
    }
    const int n = tn + l15;
    const int mb = tm + ((lane < 16) ? 0 : 8);
    #pragma unroll
    for (int r = 0; r < 8; ++r)
        C[(size_t)(mb + r) * N + n] = acc[r];
}

// ---------------- conv3 1x1 64->1 + sigmoid ----------------
__global__ void conv3_kernel(const _Float16* __restrict__ act, const float* __restrict__ w3,
                             const float* __restrict__ b3, float* __restrict__ mask) {
    int p = blockIdx.x * blockDim.x + threadIdx.x;   // B*HW
    if (p >= BATCH * HWPIX) return;
    const _Float16* a = act + ((size_t)p << 6);
    float s = b3[0];
    #pragma unroll
    for (int c = 0; c < 64; ++c) s += (float)a[c] * w3[c];
    mask[p] = 1.f / (1.f + expf(-s));
}

// ---------------- bottom-k selection: histogram -> scan -> ordered keep ----------------
__global__ void hist_kernel(const float* __restrict__ mask, int* __restrict__ hist) {
    __shared__ int lh[NBINS];
    for (int i = threadIdx.x; i < NBINS; i += blockDim.x) lh[i] = 0;
    __syncthreads();
    int b = blockIdx.x;
    for (int i = threadIdx.x; i < HWPIX; i += blockDim.x) {
        float v = mask[b * HWPIX + i];
        int bin = (int)(v * (float)NBINS);
        bin = bin < 0 ? 0 : (bin > NBINS - 1 ? NBINS - 1 : bin);
        atomicAdd(&lh[bin], 1);
    }
    __syncthreads();
    for (int i = threadIdx.x; i < NBINS; i += blockDim.x) hist[b * NBINS + i] = lh[i];
}

__global__ void scan_kernel(const int* __restrict__ hist, const int* __restrict__ kptr,
                            int* __restrict__ thr) {
    if (threadIdx.x != 0) return;
    int b = blockIdx.x;
    int kpix = (HWPIX * kptr[0]) / 100;
    int cum = 0, t = 0;
    for (; t < NBINS; ++t) {
        int h = hist[b * NBINS + t];
        if (cum + h >= kpix) break;
        cum += h;
    }
    if (t > NBINS - 1) t = NBINS - 1;
    thr[b * 2] = t;
    thr[b * 2 + 1] = kpix - cum;   // zero budget inside threshold bin
}

// one wave per sample; deterministic index-ordered tie-breaking inside the threshold bin
__global__ void keep_kernel(const float* __restrict__ mask, const int* __restrict__ thr,
                            float* __restrict__ keep) {
    int b = blockIdx.x;
    int lane = threadIdx.x;   // blockDim.x == 32
    int t = thr[b * 2];
    int budget = thr[b * 2 + 1];
    __shared__ int tflag[32];
    int consumed = 0;
    for (int base = 0; base < HWPIX; base += 32) {
        int i = base + lane;
        float v = mask[b * HWPIX + i];
        int bin = (int)(v * (float)NBINS);
        bin = bin < 0 ? 0 : (bin > NBINS - 1 ? NBINS - 1 : bin);
        bool zero = (bin < t);
        int tie = (bin == t) ? 1 : 0;
        tflag[lane] = tie;
        __syncthreads();
        int cnt_lt = 0, tot = 0;
        for (int j = 0; j < 32; ++j) { tot += tflag[j]; if (j < lane) cnt_lt += tflag[j]; }
        if (tie && (consumed + cnt_lt) < budget) zero = true;
        consumed += tot;
        keep[b * HWPIX + i] = zero ? 0.f : 1.f;
        __syncthreads();
    }
}

// filtered = x * keep -> d_out (f32)
__global__ void apply_kernel(const float* __restrict__ x, const float* __restrict__ keep,
                             float* __restrict__ outF) {
    size_t i = (size_t)blockIdx.x * blockDim.x + threadIdx.x;   // B*3*HW
    if (i >= (size_t)BATCH * 3 * HWPIX) return;
    int pix = (int)(i % HWPIX);
    int b = (int)(i / ((size_t)3 * HWPIX));
    outF[i] = x[i] * keep[b * HWPIX + pix];
}

// pack stem im2col A: [B*1024 rows][160 k] f16, k = c*49+ky*7+kx (zero pad >=147)
__global__ void packstem_kernel(const float* __restrict__ x, const float* __restrict__ keep,
                                _Float16* __restrict__ As) {
    int i = blockIdx.x * blockDim.x + threadIdx.x;   // 16384*160
    if (i >= BATCH * GHW * KSTEM) return;
    int row = i / KSTEM; int k = i - row * KSTEM;
    _Float16 v = (_Float16)0.f;
    if (k < 147) {
        int c = k / 49; int r = k - 49 * c; int ky = r / 7, kx = r % 7;
        int b = row >> 10; int pp = row & 1023;
        int py = pp >> 5; int px = pp & 31;
        int yy = py * 7 + ky, xx = px * 7 + kx;
        v = (_Float16)(x[((size_t)(b * 3 + c) * IH + yy) * IW + xx]
                       * keep[b * HWPIX + yy * IW + xx]);
    }
    As[i] = v;
}

// ---------------- mixer elementwise kernels ----------------
__global__ void bias_gelu_kernel(float* __restrict__ raw, const float* __restrict__ bias,
                                 int rows, int C) {
    size_t i = (size_t)blockIdx.x * blockDim.x + threadIdx.x;
    if (i >= (size_t)rows * C) return;
    int c = (int)(i % C);
    raw[i] = gelu_exact(raw[i] + bias[c]);
}

__global__ void bn_mix_kernel(const float* __restrict__ raw,
                              const float* __restrict__ mean, const float* __restrict__ var,
                              const float* __restrict__ g, const float* __restrict__ be,
                              float* __restrict__ h, _Float16* __restrict__ hf,
                              int rows, int C, int residual) {
    size_t i = (size_t)blockIdx.x * blockDim.x + threadIdx.x;
    if (i >= (size_t)rows * C) return;
    int c = (int)(i % C);
    float t = g[c] * (raw[i] - mean[c]) * rsqrtf(var[c] + 1e-5f) + be[c];
    if (residual) t += h[i];
    h[i] = t;
    hf[i] = (_Float16)t;
}

// depthwise 9x9 SAME on NHWC [B,32,32,256] f32 (no bias; added in bias_gelu)
__global__ void dwconv_kernel(const float* __restrict__ h, const float* __restrict__ w,
                              float* __restrict__ raw) {
    int idx = blockIdx.x * blockDim.x + threadIdx.x;   // B*1024*256
    if (idx >= BATCH * GHW * DIM) return;
    int c = idx & 255;
    int p = idx >> 8;
    int b = p >> 10; int pp = p & 1023;
    int y = pp >> 5; int x = pp & 31;
    float acc = 0.f;
    const float* wc = w + c * 81;
    for (int ky = 0; ky < 9; ++ky) {
        int yy = y + ky - 4; if ((unsigned)yy >= GH) continue;
        for (int kx = 0; kx < 9; ++kx) {
            int xx = x + kx - 4; if ((unsigned)xx >= GW) continue;
            acc += h[(((size_t)(b << 10) + (yy << 5) + xx) << 8) + c] * wc[ky * 9 + kx];
        }
    }
    raw[idx] = acc;
}

__global__ void pool_kernel(const float* __restrict__ h, float* __restrict__ pooled) {
    int b = blockIdx.x; int c = threadIdx.x;   // blockDim = 256
    float s = 0.f;
    for (int p = 0; p < GHW; ++p) s += h[(((size_t)b * GHW + p) << 8) + c];
    pooled[b * DIM + c] = s * (1.f / (float)GHW);
}

__global__ void head_kernel(const float* __restrict__ pooled, const float* __restrict__ wh,
                            const float* __restrict__ bh, float* __restrict__ out) {
    int i = blockIdx.x * blockDim.x + threadIdx.x;   // B*NCLS
    if (i >= BATCH * NCLS) return;
    int n = i % NCLS; int b = i / NCLS;
    float s = bh[n];
    for (int c = 0; c < DIM; ++c) s += pooled[b * DIM + c] * wh[n * DIM + c];
    out[i] = s;
}

// ==========================================================================================
extern "C" void kernel_launch(void* const* d_in, const int* in_sizes, int n_in,
                              void* d_out, int out_size, void* d_ws, size_t ws_size,
                              hipStream_t stream) {
    (void)in_sizes; (void)n_in; (void)out_size; (void)ws_size;
    const float* x     = (const float*)d_in[0];
    const int*   kptr  = (const int*)d_in[1];
    const float* w1    = (const float*)d_in[2];
    const float* g1    = (const float*)d_in[4];
    const float* be1   = (const float*)d_in[5];
    const float* w2    = (const float*)d_in[6];
    const float* g2    = (const float*)d_in[8];
    const float* be2   = (const float*)d_in[9];
    const float* w3    = (const float*)d_in[10];
    const float* b3    = (const float*)d_in[11];
    const float* stem_w = (const float*)d_in[12];
    const float* stem_b = (const float*)d_in[13];
    const float* stem_g = (const float*)d_in[14];
    const float* stem_be = (const float*)d_in[15];
    const float* head_w = (const float*)d_in[80];
    const float* head_b = (const float*)d_in[81];

    // -------- workspace carving (256B aligned) --------
    char* ws = (char*)d_ws;
    size_t off = 0;
    auto carve = [&](size_t bytes) -> char* {
        char* p = ws + off;
        off = (off + bytes + 255) & ~(size_t)255;
        return p;
    };
    const size_t rawBytes = (size_t)BATCH * HWPIX * HID * 4;       // 205.5 MB, reused as mixer raw
    const size_t actBytes = (size_t)BATCH * HWPIX * HID * 2;       // 102.8 MB, overlaid after conv3
    float*    rawA  = (float*)carve(rawBytes);
    char*     actRegion = carve(actBytes);
    float*    maskB = (float*)carve((size_t)BATCH * HWPIX * 4);
    float*    keepB = (float*)carve((size_t)BATCH * HWPIX * 4);
    _Float16* wbuf  = (_Float16*)carve(256 * 256 * 2);
    float*    meanB = (float*)carve(256 * 4);
    float*    varB  = (float*)carve(256 * 4);
    int*      histB = (int*)carve((size_t)BATCH * NBINS * 4);
    int*      thrB  = (int*)carve((size_t)BATCH * 2 * 4);

    _Float16* act1 = (_Float16*)actRegion;                          // live until conv3
    // overlays inside actRegion, valid after conv3:
    _Float16* Astem = (_Float16*)actRegion;                         // 16384*160 f16 = 5.24 MB
    char* ovl = actRegion + (((size_t)BATCH * GHW * KSTEM * 2 + 255) & ~(size_t)255);
    float*    hBuf  = (float*)ovl;                                  // 16.8 MB
    _Float16* hf16  = (_Float16*)(ovl + (size_t)BATCH * GHW * DIM * 4);
    float*    pooled = (float*)((char*)hf16 + (size_t)BATCH * GHW * DIM * 2);
    float*    rawMix = rawA;                                        // reuse big raw buffer

    float* outCls  = (float*)d_out;                                 // [16,1000]
    float* outFilt = (float*)d_out + BATCH * NCLS;                  // [16,3,224,224]

    const int T = 256;
    const int nAct = BATCH * HWPIX * HID;
    const int nMix = BATCH * GHW * DIM;

    // ========== filter CNN ==========
    conv1_kernel<<<(nAct + T - 1) / T, T, 0, stream>>>(x, w1, rawA);
    stats_kernel<<<HID, T, 0, stream>>>(rawA, BATCH * HWPIX, HID, meanB, varB);
    bn_relu_f16_kernel<<<(nAct + T - 1) / T, T, 0, stream>>>(rawA, meanB, varB, g1, be1,
                                                             act1, BATCH * HWPIX, HID);
    wconv2T_kernel<<<(64 * 576 + T - 1) / T, T, 0, stream>>>(w2, wbuf);
    // 50176 M-tiles * 4 N-tiles = 200704 waves / 8 per block
    gemm_conv2_wmma<<<(BATCH * HWPIX / 16) * 4 / 8, T, 0, stream>>>(act1, wbuf, rawA);
    stats_kernel<<<HID, T, 0, stream>>>(rawA, BATCH * HWPIX, HID, meanB, varB);
    bn_relu_f16_kernel<<<(nAct + T - 1) / T, T, 0, stream>>>(rawA, meanB, varB, g2, be2,
                                                             act1, BATCH * HWPIX, HID);
    conv3_kernel<<<(BATCH * HWPIX + T - 1) / T, T, 0, stream>>>(act1, w3, b3, maskB);

    // ========== bottom-k masking ==========
    hist_kernel<<<BATCH, T, 0, stream>>>(maskB, histB);
    scan_kernel<<<BATCH, 32, 0, stream>>>(histB, kptr, thrB);
    keep_kernel<<<BATCH, 32, 0, stream>>>(maskB, thrB, keepB);
    apply_kernel<<<(BATCH * 3 * HWPIX + T - 1) / T, T, 0, stream>>>(x, keepB, outFilt);

    // ========== ConvMixer stem ==========
    packstem_kernel<<<(BATCH * GHW * KSTEM + T - 1) / T, T, 0, stream>>>(x, keepB, Astem);
    wstemT_kernel<<<(256 * KSTEM + T - 1) / T, T, 0, stream>>>(stem_w, wbuf);
    const int gemmMixBlocks = (BATCH * GHW / 16) * (DIM / 16) / 8;   // 2048
    gemm_dense_wmma<<<gemmMixBlocks, T, 0, stream>>>(Astem, wbuf, rawMix,
                                                     BATCH * GHW, DIM, KSTEM);
    bias_gelu_kernel<<<(nMix + T - 1) / T, T, 0, stream>>>(rawMix, stem_b, BATCH * GHW, DIM);
    stats_kernel<<<DIM, T, 0, stream>>>(rawMix, BATCH * GHW, DIM, meanB, varB);
    bn_mix_kernel<<<(nMix + T - 1) / T, T, 0, stream>>>(rawMix, meanB, varB, stem_g, stem_be,
                                                        hBuf, hf16, BATCH * GHW, DIM, 0);

    // ========== ConvMixer blocks ==========
    for (int blk = 0; blk < DEPTH; ++blk) {
        const float* dw_w  = (const float*)d_in[16 + 8 * blk + 0];
        const float* dw_b  = (const float*)d_in[16 + 8 * blk + 1];
        const float* dw_g  = (const float*)d_in[16 + 8 * blk + 2];
        const float* dw_be = (const float*)d_in[16 + 8 * blk + 3];
        const float* pw_w  = (const float*)d_in[16 + 8 * blk + 4];
        const float* pw_b  = (const float*)d_in[16 + 8 * blk + 5];
        const float* pw_g  = (const float*)d_in[16 + 8 * blk + 6];
        const float* pw_be = (const float*)d_in[16 + 8 * blk + 7];

        dwconv_kernel<<<(nMix + T - 1) / T, T, 0, stream>>>(hBuf, dw_w, rawMix);
        bias_gelu_kernel<<<(nMix + T - 1) / T, T, 0, stream>>>(rawMix, dw_b, BATCH * GHW, DIM);
        stats_kernel<<<DIM, T, 0, stream>>>(rawMix, BATCH * GHW, DIM, meanB, varB);
        bn_mix_kernel<<<(nMix + T - 1) / T, T, 0, stream>>>(rawMix, meanB, varB, dw_g, dw_be,
                                                            hBuf, hf16, BATCH * GHW, DIM, 1);

        wpwT_kernel<<<(256 * 256 + T - 1) / T, T, 0, stream>>>(pw_w, wbuf);
        gemm_dense_wmma<<<gemmMixBlocks, T, 0, stream>>>(hf16, wbuf, rawMix,
                                                         BATCH * GHW, DIM, DIM);
        bias_gelu_kernel<<<(nMix + T - 1) / T, T, 0, stream>>>(rawMix, pw_b, BATCH * GHW, DIM);
        stats_kernel<<<DIM, T, 0, stream>>>(rawMix, BATCH * GHW, DIM, meanB, varB);
        bn_mix_kernel<<<(nMix + T - 1) / T, T, 0, stream>>>(rawMix, meanB, varB, pw_g, pw_be,
                                                            hBuf, hf16, BATCH * GHW, DIM, 0);
    }

    // ========== pool + head ==========
    pool_kernel<<<BATCH, DIM, 0, stream>>>(hBuf, pooled);
    head_kernel<<<(BATCH * NCLS + T - 1) / T, T, 0, stream>>>(pooled, head_w, head_b, outCls);
}
